// QEffMultiHeadDotProductAttention_49323404427619
// MI455X (gfx1250) — compile-verified, matrix operand order checked by reading, not run on hardware
//
#include <hip/hip_runtime.h>

#define BATCH     2
#define SEQ       2048
#define DMODEL    2048
#define NHEADS    16
#define NKVHEADS  4
#define HDIM      128

typedef __attribute__((ext_vector_type(16))) __bf16 v16bf;
typedef __attribute__((ext_vector_type(8)))  __bf16 v8bf;
typedef __attribute__((ext_vector_type(8)))  float  v8f;

__device__ __forceinline__ unsigned short f32_to_bf16(float f) {
  union { float f; unsigned u; } cv; cv.f = f;
  unsigned u = cv.u;
  u += 0x7fffu + ((u >> 16) & 1u);   // round-to-nearest-even
  return (unsigned short)(u >> 16);
}

// Load one 16-element bf16 WMMA fragment half-row: two contiguous 16B chunks
// at +0 and +16 elements (matches CDNA5 16-bit A/B per-lane K layout).
__device__ __forceinline__ v16bf load_frag(const unsigned short* p) {
  v8bf lo = *(const v8bf*)(p);
  v8bf hi = *(const v8bf*)(p + 16);
  return __builtin_shufflevector(lo, hi, 0,1,2,3,4,5,6,7,8,9,10,11,12,13,14,15);
}

__device__ __forceinline__ v8f wmma_bf16f32(v16bf a, v16bf b, v8f c) {
  return __builtin_amdgcn_wmma_f32_16x16x32_bf16(false, a, false, b, (short)0, c,
                                                 false, false);
}

// ---------------------------------------------------------------------------
// Elementwise f32 -> bf16
__global__ void cvt_f32_to_bf16(const float* __restrict__ src,
                                unsigned short* __restrict__ dst, int n) {
  int i = blockIdx.x * blockDim.x + threadIdx.x;
  if (i < n) dst[i] = f32_to_bf16(src[i]);
}

// Transpose-convert weight: src [K][N] f32 -> dst [N][K] bf16
__global__ void transpose_cvt(const float* __restrict__ src,
                              unsigned short* __restrict__ dst, int K, int N) {
  int i = blockIdx.x * blockDim.x + threadIdx.x;
  if (i < K * N) {
    int k = i / N, n = i - k * N;
    dst[(size_t)n * K + k] = f32_to_bf16(src[i]);
  }
}

// ---------------------------------------------------------------------------
// GEMM: C[M][N] = (A[M][K] @ Bt[N][K]^T + bias[N]) * scale
// A, Bt are bf16; C is bf16 (projections) or f32 (final output).
// Tile 128x128x32, 256 threads = 8 waves (4 in M x 2 in N), each wave owns
// a 32x64 region = 2x4 WMMA 16x16 accumulators.
template <bool OUT_BF16>
__global__ void __launch_bounds__(256)
gemm_xw(const unsigned short* __restrict__ A,
        const unsigned short* __restrict__ Bt,
        const float* __restrict__ bias,
        void* __restrict__ C,
        int M, int N, int K, float scale) {
  constexpr int BM = 128, BN = 128, BK = 32, LDT = BK + 8;  // 80B rows, 16B-aligned
  __shared__ unsigned short As[BM * LDT];
  __shared__ unsigned short Bs[BN * LDT];

  const int tid  = threadIdx.x;
  const int wid  = tid >> 5, lane = tid & 31;
  const int wm   = wid & 3,  wn   = wid >> 2;
  const int hgrp = lane >> 4, r   = lane & 15;
  const int bm   = blockIdx.y * BM, bn = blockIdx.x * BN;
  const int cb   = hgrp * 8;

  v8f acc[2][4] = {};

  for (int k0 = 0; k0 < K; k0 += BK) {
    __syncthreads();
    // Stage A and B tiles: 128x32 bf16 each = 512 uint4; 2 per thread.
#pragma unroll
    for (int it = 0; it < 2; ++it) {
      int idx = tid + it * 256;
      int row = idx >> 2;
      int c8  = (idx & 3) * 8;
      *(uint4*)&As[row * LDT + c8] = *(const uint4*)&A [(size_t)(bm + row) * K + k0 + c8];
      *(uint4*)&Bs[row * LDT + c8] = *(const uint4*)&Bt[(size_t)(bn + row) * K + k0 + c8];
    }
    if (k0 + BK < K) {  // global_prefetch_b8 of next k-slab
      __builtin_prefetch(&A [(size_t)(bm + (tid & 127)) * K + k0 + BK], 0, 0);
      __builtin_prefetch(&Bt[(size_t)(bn + (tid & 127)) * K + k0 + BK], 0, 0);
    }
    __syncthreads();

    v16bf a0 = load_frag(&As[(wm * 32 +      r) * LDT + cb]);
    v16bf a1 = load_frag(&As[(wm * 32 + 16 + r) * LDT + cb]);
#pragma unroll
    for (int tn = 0; tn < 4; ++tn) {
      v16bf bf = load_frag(&Bs[(wn * 64 + tn * 16 + r) * LDT + cb]);
      acc[0][tn] = wmma_bf16f32(a0, bf, acc[0][tn]);
      acc[1][tn] = wmma_bf16f32(a1, bf, acc[1][tn]);
    }
  }

  // Epilogue. C/D layout: VGPR i, lanes 0-15 -> M=i, lanes 16-31 -> M=i+8.
#pragma unroll
  for (int tn = 0; tn < 4; ++tn) {
    int col = bn + wn * 64 + tn * 16 + r;
    float bv = bias[col];
#pragma unroll
    for (int tm = 0; tm < 2; ++tm) {
#pragma unroll
      for (int i = 0; i < 8; ++i) {
        int row = bm + wm * 32 + tm * 16 + i + 8 * hgrp;
        float v = (acc[tm][tn][i] + bv) * scale;
        if (OUT_BF16)
          ((unsigned short*)C)[(size_t)row * N + col] = f32_to_bf16(v);
        else
          ((float*)C)[(size_t)row * N + col] = v;
      }
    }
  }
}

// ---------------------------------------------------------------------------
// Flash attention (GQA). Grid: (SEQ/128, BATCH*NHEADS), 256 threads.
// Each wave owns 16 q-rows; all 8 waves share 32-key K/V LDS tiles.
// Q is pre-scaled by 1/sqrt(D) at projection time.
__global__ void __launch_bounds__(256)
flash_attn(const unsigned short* __restrict__ Q,   // [B,S,H,D]  bf16
           const unsigned short* __restrict__ Kg,  // [B,S,Hk,D] bf16
           const unsigned short* __restrict__ Vg,  // [B,S,Hk,D] bf16
           unsigned short* __restrict__ O) {       // [B,S,H,D]  bf16
  constexpr int Sq = SEQ, H = NHEADS, HK = NKVHEADS, D = HDIM, TK = 32;
  __shared__ unsigned short Ks[TK * D];       // [key][d]   8 KB
  __shared__ unsigned short Vs[D * TK];       // [d][key]   8 KB (transposed)
  __shared__ unsigned short Ps[8 * 16 * TK];  // per-wave P 8 KB

  const int tid  = threadIdx.x, wid = tid >> 5, lane = tid & 31;
  const int hgrp = lane >> 4,  r   = lane & 15;
  const int bh   = blockIdx.y, b = bh / H, h = bh % H, hk = h >> 2;  // G=4
  const int q0   = blockIdx.x * 128 + wid * 16;

  // Q fragments: 4 x (16x32) A-matrices covering D=128.
  const unsigned short* qrow = Q + ((size_t)(b * Sq + q0 + r) * H + h) * D;
  v16bf qf[4];
#pragma unroll
  for (int t = 0; t < 4; ++t) qf[t] = load_frag(qrow + t * 32 + hgrp * 8);

  float mrow[8], lrow[8];
  v8f acc[8] = {};
#pragma unroll
  for (int i = 0; i < 8; ++i) { mrow[i] = -3.0e38f; lrow[i] = 0.f; }

  unsigned short* pw = &Ps[wid * 16 * TK];

  for (int kb = 0; kb < Sq; kb += TK) {
    __syncthreads();
    // Stage K (row-major) and V (transposed) tiles: 512 uint4 each, 2/thread.
#pragma unroll
    for (int it = 0; it < 2; ++it) {
      int idx = tid + it * 256;
      int key = idx >> 4, d0 = (idx & 15) * 8;
      size_t gbase = ((size_t)(b * Sq + kb + key) * HK + hk) * D + d0;
      uint4 kv = *(const uint4*)&Kg[gbase];
      *(uint4*)&Ks[key * D + d0] = kv;
      uint4 vv = *(const uint4*)&Vg[gbase];
      const unsigned short* pv = (const unsigned short*)&vv;
#pragma unroll
      for (int j = 0; j < 8; ++j) Vs[(d0 + j) * TK + key] = pv[j];
    }
    __syncthreads();

    // Scores: two 16-key tiles, accumulate over 4 d-chunks.
    v8f s0 = {}, s1 = {};
#pragma unroll
    for (int t = 0; t < 4; ++t) {
      v16bf k0f = load_frag(&Ks[(     r) * D + t * 32 + hgrp * 8]);
      v16bf k1f = load_frag(&Ks[(16 + r) * D + t * 32 + hgrp * 8]);
      s0 = wmma_bf16f32(qf[t], k0f, s0);
      s1 = wmma_bf16f32(qf[t], k1f, s1);
    }

    // Online softmax: row stats live replicated across each 16-lane half.
    float alpha[8], p0[8], p1[8];
#pragma unroll
    for (int i = 0; i < 8; ++i) {
      float mx = fmaxf(s0[i], s1[i]);
#pragma unroll
      for (int msk = 1; msk < 16; msk <<= 1)
        mx = fmaxf(mx, __shfl_xor(mx, msk, 32));
      float mnew = fmaxf(mrow[i], mx);
      alpha[i] = __expf(mrow[i] - mnew);
      p0[i] = __expf(s0[i] - mnew);
      p1[i] = __expf(s1[i] - mnew);
      float sum = p0[i] + p1[i];
#pragma unroll
      for (int msk = 1; msk < 16; msk <<= 1)
        sum += __shfl_xor(sum, msk, 32);
      lrow[i] = lrow[i] * alpha[i] + sum;
      mrow[i] = mnew;
    }
#pragma unroll
    for (int c = 0; c < 8; ++c)
#pragma unroll
      for (int i = 0; i < 8; ++i) acc[c][i] *= alpha[i];

    // Rearrange P (C-layout) -> A-layout via per-wave LDS (wave-local DS ops
    // are in-order, no block barrier needed).
#pragma unroll
    for (int i = 0; i < 8; ++i) {
      int m = i + 8 * hgrp;
      pw[m * TK +      r] = f32_to_bf16(p0[i]);
      pw[m * TK + 16 + r] = f32_to_bf16(p1[i]);
    }
    __builtin_amdgcn_wave_barrier();
    v16bf pa = load_frag(&pw[r * TK + hgrp * 8]);

    // acc += P @ V over 8 d-chunks (V transposed in LDS -> contiguous K).
#pragma unroll
    for (int c = 0; c < 8; ++c) {
      v16bf vf = load_frag(&Vs[(c * 16 + r) * TK + hgrp * 8]);
      acc[c] = wmma_bf16f32(pa, vf, acc[c]);
    }
  }

  // Normalize and store context O = acc / l.
  float linv[8];
#pragma unroll
  for (int i = 0; i < 8; ++i) linv[i] = 1.0f / lrow[i];
#pragma unroll
  for (int c = 0; c < 8; ++c) {
#pragma unroll
    for (int i = 0; i < 8; ++i) {
      int row = q0 + i + 8 * hgrp;
      O[((size_t)(b * Sq + row) * H + h) * D + c * 16 + r] =
          f32_to_bf16(acc[c][i] * linv[i]);
    }
  }
}

// ---------------------------------------------------------------------------
extern "C" void kernel_launch(void* const* d_in, const int* in_sizes, int n_in,
                              void* d_out, int out_size, void* d_ws, size_t ws_size,
                              hipStream_t stream) {
  const float* x  = (const float*)d_in[0];
  const float* wq = (const float*)d_in[1];
  const float* bq = (const float*)d_in[2];
  const float* wk = (const float*)d_in[3];
  const float* bk = (const float*)d_in[4];
  const float* wv = (const float*)d_in[5];
  const float* bv = (const float*)d_in[6];
  const float* wo = (const float*)d_in[7];
  const float* bo = (const float*)d_in[8];
  float* out = (float*)d_out;
  (void)in_sizes; (void)n_in; (void)out_size; (void)ws_size;

  const int M   = BATCH * SEQ;        // 4096
  const int NQ  = NHEADS * HDIM;      // 2048
  const int NKV = NKVHEADS * HDIM;    // 512

  char* ws = (char*)d_ws;
  size_t off = 0;
  auto take = [&](size_t elems) {
    unsigned short* p = (unsigned short*)(ws + off);
    off += elems * sizeof(unsigned short);
    off = (off + 255) & ~(size_t)255;
    return p;
  };
  unsigned short* Xb  = take((size_t)M * DMODEL);
  unsigned short* Wqt = take((size_t)NQ * DMODEL);
  unsigned short* Wkt = take((size_t)NKV * DMODEL);
  unsigned short* Wvt = take((size_t)NKV * DMODEL);
  unsigned short* Wot = take((size_t)DMODEL * NQ);
  unsigned short* Qb  = take((size_t)M * NQ);
  unsigned short* Kb  = take((size_t)M * NKV);
  unsigned short* Vb  = take((size_t)M * NKV);
  unsigned short* Ob  = take((size_t)M * NQ);

  const int T = 256;
  int nX = M * DMODEL;
  cvt_f32_to_bf16<<<(nX + T - 1) / T, T, 0, stream>>>(x, Xb, nX);
  transpose_cvt<<<(DMODEL * NQ  + T - 1) / T, T, 0, stream>>>(wq, Wqt, DMODEL, NQ);
  transpose_cvt<<<(DMODEL * NKV + T - 1) / T, T, 0, stream>>>(wk, Wkt, DMODEL, NKV);
  transpose_cvt<<<(DMODEL * NKV + T - 1) / T, T, 0, stream>>>(wv, Wvt, DMODEL, NKV);
  transpose_cvt<<<(NQ * DMODEL  + T - 1) / T, T, 0, stream>>>(wo, Wot, NQ, DMODEL);

  const float qscale = 0.088388347648318447f;  // 1/sqrt(128)
  gemm_xw<true ><<<dim3(NQ  / 128, M / 128), 256, 0, stream>>>(Xb, Wqt, bq, Qb, M, NQ,  DMODEL, qscale);
  gemm_xw<true ><<<dim3(NKV / 128, M / 128), 256, 0, stream>>>(Xb, Wkt, bk, Kb, M, NKV, DMODEL, 1.0f);
  gemm_xw<true ><<<dim3(NKV / 128, M / 128), 256, 0, stream>>>(Xb, Wvt, bv, Vb, M, NKV, DMODEL, 1.0f);

  flash_attn<<<dim3(SEQ / 128, BATCH * NHEADS), 256, 0, stream>>>(Qb, Kb, Vb, Ob);

  gemm_xw<false><<<dim3(DMODEL / 128, M / 128), 256, 0, stream>>>(Ob, Wot, bo, out, M, DMODEL, NQ, 1.0f);
}
